// Attention_3092376453647
// MI455X (gfx1250) — compile-verified, matrix operand order checked by reading, not run on hardware
//
#include <hip/hip_runtime.h>

// ---------------- types & WMMA helpers ----------------
typedef _Float16 half_t;
typedef __attribute__((ext_vector_type(16))) _Float16 v16h;
typedef __attribute__((ext_vector_type(8)))  float    v8f;

#define WMMA_F16(a, b, c) \
  __builtin_amdgcn_wmma_f32_16x16x32_f16(false, (a), false, (b), (short)0, (c), false, false)

static __device__ __forceinline__ v8f zero8() {
  v8f z;
#pragma unroll
  for (int i = 0; i < 8; ++i) z[i] = 0.0f;
  return z;
}

// Load a 16x32 fp16 fragment (A layout; also used for B since the layout is
// the mirror with "row" meaning the 16 columns of B). `base` points at
// element (0,0); rows are strided by `ld` halves, K is contiguous.
// Per ISA 7.12.2 (16-bit 16x32): lane m=lane&15 holds row m; half-group lo
// holds K = {k0..k0+7, k0+16..k0+23}, half-group hi holds K+8 of each.
static __device__ __forceinline__ v16h load_frag(const half_t* base, int ld,
                                                 int row0, int k0, int lane) {
  const int m  = lane & 15;
  const int hi = lane >> 4;
  const half_t* p = base + ((size_t)(row0 + m)) * (size_t)ld + (size_t)(k0 + hi * 8);
  v16h f;
#pragma unroll
  for (int i = 0; i < 8; ++i) { f[i] = p[i]; f[i + 8] = p[i + 16]; }
  return f;
}

// ---------------- problem constants ----------------
#define BB 2
#define CC 256
#define NN 4096
#define HEADS 4
#define DH 64
#define INNER 256
#define SCALE 8.0f

// ---------------- kernel 0: fp32 -> fp16 weight conversion ----------------
__global__ void k_convert(const float* __restrict__ wqkv, const float* __restrict__ wout,
                          half_t* __restrict__ wqkv_h, half_t* __restrict__ wout_h) {
  int i = blockIdx.x * 256 + threadIdx.x;
  if (i < 768 * 256) wqkv_h[i] = (half_t)wqkv[i];
  if (i < 256 * 256) wout_h[i] = (half_t)wout[i];
}

// ---------------- kernel 1: LayerNorm + QKV projection (WMMA) ----------------
// grid: 512 blocks = (b, row-half, 32-token tile), 256 threads (8 waves).
// Each block computes 384 output rows x 32 cols; wave w owns 3x2 tiles
// (48 accumulator VGPRs -> no spills).
__global__ void k_ln_qkv(const float* __restrict__ x, const float* __restrict__ gamma,
                         const half_t* __restrict__ wq, half_t* __restrict__ Q,
                         half_t* __restrict__ K, half_t* __restrict__ vT) {
  __shared__ half_t xns[32][264];   // [token col][channel], padded stride
  __shared__ float  sS[8][32], sQ2[8][32];
  __shared__ float  smean[32], sinv[32];

  const int tid   = threadIdx.x;
  const int b     = blockIdx.x >> 8;
  const int rhalf = (blockIdx.x >> 7) & 1;     // which 384-row half of QKV
  const int n0    = (blockIdx.x & 127) << 5;
  const int col   = tid & 31;
  const int part  = tid >> 5;
  const int n     = n0 + col;
  const float* xb = x + (size_t)b * CC * NN;

  // ---- LayerNorm stats over C=256 (8 partial sums per token) ----
  float s = 0.0f, q2 = 0.0f;
#pragma unroll 4
  for (int c0 = 0; c0 < 32; ++c0) {
    int c = part * 32 + c0;
    float v = xb[(size_t)c * NN + n];
    s += v; q2 += v * v;
  }
  sS[part][col] = s; sQ2[part][col] = q2;
  __syncthreads();
  if (part == 0) {
    float ts = 0.0f, tq = 0.0f;
#pragma unroll
    for (int p = 0; p < 8; ++p) { ts += sS[p][col]; tq += sQ2[p][col]; }
    float mean = ts * (1.0f / 256.0f);
    float var  = tq * (1.0f / 256.0f) - mean * mean;
    smean[col] = mean;
    sinv[col]  = rsqrtf(var + 1e-5f);
  }
  __syncthreads();
  {
    float mean = smean[col], inv = sinv[col];
#pragma unroll 4
    for (int c0 = 0; c0 < 32; ++c0) {
      int c = part * 32 + c0;
      float v = (xb[(size_t)c * NN + n] - mean) * inv * gamma[c];
      xns[col][c] = (half_t)v;
    }
  }
  __syncthreads();

  // ---- GEMM: [384x256] x [256x32] with v_wmma_f32_16x16x32_f16 ----
  const int lane = tid & 31;
  const int w    = tid >> 5;
  const int rbase = rhalf * 384 + w * 48;      // first output row of this wave
  v8f acc[3][2];
#pragma unroll
  for (int rt = 0; rt < 3; ++rt) { acc[rt][0] = zero8(); acc[rt][1] = zero8(); }

#pragma unroll 1
  for (int k0 = 0; k0 < 256; k0 += 32) {
    v16h bf0 = load_frag(&xns[0][0], 264, 0,  k0, lane);   // B: columns n0..n0+15
    v16h bf1 = load_frag(&xns[0][0], 264, 16, k0, lane);   // B: columns n0+16..n0+31
#pragma unroll
    for (int rt = 0; rt < 3; ++rt) {
      v16h af = load_frag(wq, 256, rbase + rt * 16, k0, lane);
      acc[rt][0] = WMMA_F16(af, bf0, acc[rt][0]);
      acc[rt][1] = WMMA_F16(af, bf1, acc[rt][1]);
    }
  }

  // ---- scatter epilogue into Q / K / vT ----
  const int hi = lane >> 4, ln = lane & 15;
#pragma unroll
  for (int rt = 0; rt < 3; ++rt) {
#pragma unroll
    for (int ct = 0; ct < 2; ++ct) {
#pragma unroll
      for (int r = 0; r < 8; ++r) {
        int o  = rbase + rt * 16 + r + 8 * hi;   // 0..767
        int nn = n0 + ct * 16 + ln;
        half_t hv = (half_t)acc[rt][ct][r];
        int which = o >> 8, rem = o & 255;
        int h = rem >> 6, d = rem & 63;
        size_t bh = (size_t)(b * HEADS + h);
        if (which == 0)      Q[(bh * NN + nn) * DH + d]  = hv;
        else if (which == 1) K[(bh * NN + nn) * DH + d]  = hv;
        else                 vT[(bh * DH + d) * NN + nn] = hv;
      }
    }
  }
}

// ---------------- kernel 2: L2 normalize q and k rows ----------------
__global__ void k_l2norm(half_t* __restrict__ Q, half_t* __restrict__ K) {
  int idx   = blockIdx.x * 256 + threadIdx.x;       // 0..65535
  int which = idx >> 15;                            // 32768 rows per tensor
  int row   = idx & 32767;
  half_t* p = (which ? K : Q) + (size_t)row * DH;
  float s = 0.0f;
#pragma unroll 8
  for (int i = 0; i < DH; ++i) { float v = (float)p[i]; s += v * v; }
  float inv = rsqrtf(s + 1e-12f);
#pragma unroll 8
  for (int i = 0; i < DH; ++i) p[i] = (half_t)((float)p[i] * inv);
}

// ---------------- kernel 3: flash attention (WMMA) ----------------
// grid: 256 blocks x 256 threads; each wave owns one 16-query tile of one
// (b, h) and streams all 4096 keys in steps of 32 with online softmax.
__global__ void k_flash(const half_t* __restrict__ Q, const half_t* __restrict__ K,
                        const half_t* __restrict__ vT, half_t* __restrict__ AO) {
  __shared__ half_t pbuf[8][16][40];   // per-wave P staging tile (padded)

  const int tid  = threadIdx.x;
  const int lane = tid & 31;
  const int w    = tid >> 5;
  const int bh   = blockIdx.x >> 5;          // 0..7
  const int grp  = blockIdx.x & 31;
  const int m0   = (grp * 8 + w) * 16;       // query tile base
  const int b    = bh >> 2, h = bh & 3;
  const int hi = lane >> 4, ln = lane & 15;

  const half_t* qp = Q  + (size_t)bh * NN * DH;
  const half_t* kp = K  + (size_t)bh * NN * DH;
  const half_t* vp = vT + (size_t)bh * DH * NN;

  // Q A-fragments (resident in VGPRs)
  v16h qa0 = load_frag(qp, DH, m0, 0,  lane);
  v16h qa1 = load_frag(qp, DH, m0, 32, lane);

  float rmax[8], rsum[8];
  v8f acc[4];
#pragma unroll
  for (int r = 0; r < 8; ++r) { rmax[r] = -3.0e38f; rsum[r] = 0.0f; }
#pragma unroll
  for (int dt = 0; dt < 4; ++dt) acc[dt] = zero8();

#pragma unroll 1
  for (int j0 = 0; j0 < NN; j0 += 32) {
    if (j0 + 64 < NN) {
      __builtin_prefetch(kp + (size_t)(j0 + 64) * DH, 0, 1);
      __builtin_prefetch(vp + (size_t)(j0 + 64), 0, 1);
    }
    // S = Q K^T  (two 16-key tiles)
    v8f s0 = zero8(), s1 = zero8();
    s0 = WMMA_F16(qa0, load_frag(kp, DH, j0,       0,  lane), s0);
    s0 = WMMA_F16(qa1, load_frag(kp, DH, j0,       32, lane), s0);
    s1 = WMMA_F16(qa0, load_frag(kp, DH, j0 + 16,  0,  lane), s1);
    s1 = WMMA_F16(qa1, load_frag(kp, DH, j0 + 16,  32, lane), s1);

    // online softmax (row reductions across 16-lane half-groups)
    float fac[8];
#pragma unroll
    for (int r = 0; r < 8; ++r) {
      float a = s0[r] * SCALE, c = s1[r] * SCALE;
      float m = fmaxf(a, c);
      m = fmaxf(m, __shfl_xor(m, 1, 32));
      m = fmaxf(m, __shfl_xor(m, 2, 32));
      m = fmaxf(m, __shfl_xor(m, 4, 32));
      m = fmaxf(m, __shfl_xor(m, 8, 32));
      float nm = fmaxf(rmax[r], m);
      float f  = __expf(rmax[r] - nm);
      rmax[r] = nm; fac[r] = f;
      float p0 = __expf(a - nm), p1 = __expf(c - nm);
      s0[r] = p0; s1[r] = p1;
      float ps = p0 + p1;
      ps += __shfl_xor(ps, 1, 32);
      ps += __shfl_xor(ps, 2, 32);
      ps += __shfl_xor(ps, 4, 32);
      ps += __shfl_xor(ps, 8, 32);
      rsum[r] = rsum[r] * f + ps;
    }
#pragma unroll
    for (int dt = 0; dt < 4; ++dt)
#pragma unroll
      for (int r = 0; r < 8; ++r) acc[dt][r] *= fac[r];

    // stage P (C/D layout -> row-major 16x32 in LDS), then re-read as A frag
#pragma unroll
    for (int r = 0; r < 8; ++r) {
      pbuf[w][r + 8 * hi][ln]      = (half_t)s0[r];
      pbuf[w][r + 8 * hi][16 + ln] = (half_t)s1[r];
    }
    v16h pa;
    {
      const half_t* pp = &pbuf[w][ln][hi * 8];
#pragma unroll
      for (int i = 0; i < 8; ++i) { pa[i] = pp[i]; pa[i + 8] = pp[i + 16]; }
    }

    // O += P V  (V^T gives contiguous keys per d column)
#pragma unroll
    for (int dt = 0; dt < 4; ++dt)
      acc[dt] = WMMA_F16(pa, load_frag(vp, NN, dt * 16, j0, lane), acc[dt]);
  }

  // epilogue: divide by row sum, write [b][query][h*64+d] fp16
#pragma unroll
  for (int r = 0; r < 8; ++r) {
    float inv = 1.0f / rsum[r];
    int qrow = m0 + r + 8 * hi;
#pragma unroll
    for (int dt = 0; dt < 4; ++dt) {
      int d = dt * 16 + ln;
      AO[((size_t)b * NN + qrow) * INNER + h * DH + d] = (half_t)(acc[dt][r] * inv);
    }
  }
}

// ---------------- kernel 4: output projection + residual (WMMA) ----------------
// grid: 256 blocks (b, 32-token tile) x 256 threads; wave owns 2x2 tiles.
__global__ void k_outproj(const half_t* __restrict__ AO, const half_t* __restrict__ wo,
                          const float* __restrict__ x, float* __restrict__ out) {
  const int tid  = threadIdx.x;
  const int lane = tid & 31;
  const int w    = tid >> 5;
  const int b    = blockIdx.x >> 7;
  const int n0   = (blockIdx.x & 127) << 5;

  v8f acc[2][2];
#pragma unroll
  for (int rt = 0; rt < 2; ++rt) { acc[rt][0] = zero8(); acc[rt][1] = zero8(); }

  const half_t* cb = AO + ((size_t)b * NN + n0) * INNER;
#pragma unroll 1
  for (int k0 = 0; k0 < 256; k0 += 32) {
    v16h bf0 = load_frag(cb, INNER, 0,  k0, lane);
    v16h bf1 = load_frag(cb, INNER, 16, k0, lane);
#pragma unroll
    for (int rt = 0; rt < 2; ++rt) {
      v16h af = load_frag(wo, 256, (w * 2 + rt) * 16, k0, lane);
      acc[rt][0] = WMMA_F16(af, bf0, acc[rt][0]);
      acc[rt][1] = WMMA_F16(af, bf1, acc[rt][1]);
    }
  }

  const int hi = lane >> 4, ln = lane & 15;
#pragma unroll
  for (int rt = 0; rt < 2; ++rt)
#pragma unroll
    for (int ct = 0; ct < 2; ++ct)
#pragma unroll
      for (int r = 0; r < 8; ++r) {
        int o = (w * 2 + rt) * 16 + r + 8 * hi;
        int n = n0 + ct * 16 + ln;
        size_t idx = ((size_t)b * CC + o) * NN + n;
        out[idx] = acc[rt][ct][r] + x[idx];
      }
}

// ---------------- host launcher ----------------
extern "C" void kernel_launch(void* const* d_in, const int* in_sizes, int n_in,
                              void* d_out, int out_size, void* d_ws, size_t ws_size,
                              hipStream_t stream) {
  const float* x     = (const float*)d_in[0];
  const float* gamma = (const float*)d_in[1];
  const float* wqkv  = (const float*)d_in[2];
  const float* wout  = (const float*)d_in[3];
  float* out = (float*)d_out;

  char* ws = (char*)d_ws;
  half_t* wqkv_h = (half_t*)(ws + 0);              //  768*256*2 = 393216
  half_t* wout_h = (half_t*)(ws + 393216);         //  256*256*2 = 131072
  half_t* Q      = (half_t*)(ws + 524288);         //  2*4*4096*64*2 = 4 MiB
  half_t* K      = (half_t*)(ws + 4718592);        //  4 MiB
  half_t* vT     = (half_t*)(ws + 8912896);        //  4 MiB
  half_t* AO     = (half_t*)(ws + 13107200);       //  2*4096*256*2 = 4 MiB

  k_convert<<<768, 256, 0, stream>>>(wqkv, wout, wqkv_h, wout_h);
  k_ln_qkv <<<512, 256, 0, stream>>>(x, gamma, wqkv_h, Q, K, vT);
  k_l2norm <<<256, 256, 0, stream>>>(Q, K);
  k_flash  <<<256, 256, 0, stream>>>(Q, K, vT, AO);
  k_outproj<<<256, 256, 0, stream>>>(AO, wout_h, x, out);
}
